// FeedForward_53807350284604
// MI455X (gfx1250) — matrix operand order repaired and checked by reading, not compiled, not run on hardware
//
#include <hip/hip_runtime.h>
#include <hip/hip_bf16.h>

// ---------------------------------------------------------------------------
// Types matching CDNA5 WMMA builtin signatures (wave32, 16x16x32 bf16).
// ---------------------------------------------------------------------------
typedef __attribute__((ext_vector_type(16))) __bf16        v16bf;
typedef __attribute__((ext_vector_type(8)))  float         v8f;
typedef __attribute__((ext_vector_type(4))) unsigned int   u32x4;
typedef __attribute__((ext_vector_type(2))) unsigned int   u32x2;

union Frag { u32x4 q[2]; v16bf v; };

__device__ inline unsigned int f2bf_u32(float f) {
  unsigned int u = __float_as_uint(f);
  u += 0x7FFFu + ((u >> 16) & 1u);   // round-to-nearest-even
  return u >> 16;
}
__device__ inline unsigned short f2bf(float f) { return (unsigned short)f2bf_u32(f); }
__device__ inline unsigned int pack2(float lo, float hi) {
  return f2bf_u32(lo) | (f2bf_u32(hi) << 16);
}

// Branch-free tanh-form GELU via hardware transcendentals (v_exp/v_rcp).
// 0.5*v*(1+tanh(0.79788456*(v+0.044715 v^3))) == v * sigmoid(2u)
__device__ inline float gelu_fast(float v) {
  float u = v * __fmaf_rn(0.0356774081f, v * v, 0.79788456f);
  return __fdividef(v, 1.0f + __expf(-2.0f * u));
}

#define GN_EPS 1e-5f
#define BS 40   // LDS row stride in ushorts (32 data + 8 pad; 80B, 16B-aligned)

// ---------------------------------------------------------------------------
// Kernel 1: GroupNorm stats. One block per (b,g): contiguous 64*4096 chunk.
// grid = 128, block = 256
// ---------------------------------------------------------------------------
__global__ __launch_bounds__(256) void gn_stats_kernel(const float* __restrict__ x,
                                                       float* __restrict__ mean,
                                                       float* __restrict__ rstd) {
  __shared__ float s_sum[256];
  __shared__ float s_sq[256];
  const int tid = threadIdx.x;
  const long base = (long)blockIdx.x * 262144;   // 64*4096
  const float4* p = (const float4*)(x + base);
  float s = 0.f, q = 0.f;
  for (int i = tid; i < 65536; i += 256) {
    float4 v = p[i];
    s += v.x + v.y + v.z + v.w;
    q += v.x * v.x + v.y * v.y + v.z * v.z + v.w * v.w;
  }
  s_sum[tid] = s; s_sq[tid] = q;
  __syncthreads();
  for (int off = 128; off > 0; off >>= 1) {
    if (tid < off) { s_sum[tid] += s_sum[tid + off]; s_sq[tid] += s_sq[tid + off]; }
    __syncthreads();
  }
  if (tid == 0) {
    float m   = s_sum[0] * (1.0f / 262144.f);
    float var = s_sq[0] * (1.0f / 262144.f) - m * m;
    mean[blockIdx.x] = m;
    rstd[blockIdx.x] = rsqrtf(var + GN_EPS);
  }
}

// ---------------------------------------------------------------------------
// Kernel 2: convert weights to bf16 once. grid = 2048, block = 256.
// ---------------------------------------------------------------------------
__global__ __launch_bounds__(256) void cvt_w_kernel(const float* __restrict__ w1,
                                                    const float* __restrict__ w2,
                                                    unsigned short* __restrict__ w1b,
                                                    unsigned short* __restrict__ w2b) {
  int i = blockIdx.x * 256 + threadIdx.x;   // 0 .. 524287
  w1b[i] = f2bf(w1[i]);
  w2b[i] = f2bf(w2[i]);
}

// ---------------------------------------------------------------------------
// Kernel 3: GEMM1  h = GELU(w1 @ groupnorm(x) + b1)   [per batch]
//   Block tile 128(M) x 256(N), 8 waves in 2x4, wave tile 64x64 (4x4 WMMA).
//   GroupNorm fused into the B-tile load (f32 -> normalized bf16 pairs).
// grid = (L/256=16, M/128=8, B=16), block = 256
// ---------------------------------------------------------------------------
__global__ __launch_bounds__(256) void gemm1_gn_gelu_kernel(
    const float* __restrict__ x,
    const unsigned short* __restrict__ w1b,
    const float* __restrict__ b1,
    const float* __restrict__ gamma,
    const float* __restrict__ beta,
    const float* __restrict__ mean,
    const float* __restrict__ rstd,
    unsigned short* __restrict__ h) {
  __shared__ unsigned short ldsA[128 * BS];   // [M][K]
  __shared__ unsigned short ldsB[256 * BS];   // [N][K] transposed for frag reads
  const int tid  = threadIdx.x;
  const int lane = tid & 31, wave = tid >> 5;
  const int lo = lane & 15, hi = lane >> 4;
  const int wm = (wave & 1) * 64;
  const int wn = (wave >> 1) * 64;
  const int l0 = blockIdx.x * 256;
  const int m0 = blockIdx.y * 128;
  const int b  = blockIdx.z;
  const long xbase = (long)b * 512 * 4096;

  v8f acc[4][4] = {};

  for (int k0 = 0; k0 < 512; k0 += 32) {
    __syncthreads();
    // --- A tile: 128x32 bf16 from w1b (ld=512), 2x b128 per thread ---
#pragma unroll
    for (int t = 0; t < 2; ++t) {
      int ch  = tid + t * 256;
      int row = ch >> 2;
      int kc  = (ch & 3) * 8;
      u32x4 v = *(const u32x4*)(w1b + (long)(m0 + row) * 512 + k0 + kc);
      *(u32x4*)(&ldsA[row * BS + kc]) = v;
    }
    // --- B tile: 32 ch x 256 cols, GN fused; channel-pair packed b32 stores
#pragma unroll
    for (int t = 0; t < 4; ++t) {
      int u  = tid + t * 256;            // unit: (channel pair, l-quad)
      int cp = u >> 6;                   // 0..15  -> channels 2cp, 2cp+1
      int l  = (u & 63) * 4;             // 0..252
      int c0 = k0 + 2 * cp;
      int sg = b * 8 + (c0 >> 6);        // both channels in same group
      float mu = mean[sg], rs = rstd[sg];
      float sc0 = rs * gamma[c0],     sh0 = beta[c0]     - mu * sc0;
      float sc1 = rs * gamma[c0 + 1], sh1 = beta[c0 + 1] - mu * sc1;
      float4 x0 = *(const float4*)(x + xbase + (long)c0 * 4096 + l0 + l);
      float4 x1 = *(const float4*)(x + xbase + (long)(c0 + 1) * 4096 + l0 + l);
      unsigned int* pb = (unsigned int*)&ldsB[l * BS + 2 * cp];
      pb[(0 * BS) / 2] = pack2(x0.x * sc0 + sh0, x1.x * sc1 + sh1);
      pb[(1 * BS) / 2] = pack2(x0.y * sc0 + sh0, x1.y * sc1 + sh1);
      pb[(2 * BS) / 2] = pack2(x0.z * sc0 + sh0, x1.z * sc1 + sh1);
      pb[(3 * BS) / 2] = pack2(x0.w * sc0 + sh0, x1.w * sc1 + sh1);
    }
    __syncthreads();
    // --- fragments: A lane m=lo, K=hi*8..+7 / 16+hi*8..+7;
    //     B lane n=lo, K=hi*8..+7 / 16+hi*8..+7 ---
    Frag fb[4];
#pragma unroll
    for (int j = 0; j < 4; ++j) {
      const unsigned short* pb = &ldsB[(wn + j * 16 + lo) * BS + hi * 8];
      fb[j].q[0] = *(const u32x4*)(pb);
      fb[j].q[1] = *(const u32x4*)(pb + 16);
    }
#pragma unroll
    for (int i = 0; i < 4; ++i) {
      Frag fa;
      const unsigned short* pa = &ldsA[(wm + i * 16 + lo) * BS + hi * 8];
      fa.q[0] = *(const u32x4*)(pa);
      fa.q[1] = *(const u32x4*)(pa + 16);
#pragma unroll
      for (int j = 0; j < 4; ++j) {
        acc[i][j] = __builtin_amdgcn_wmma_f32_16x16x32_bf16(
            false, fa.v, false, fb[j].v, (short)0, acc[i][j], false, false);
      }
    }
  }
  // --- epilogue: + b1, fast GELU, store bf16 h[b, m, l] ---
#pragma unroll
  for (int i = 0; i < 4; ++i) {
#pragma unroll
    for (int j = 0; j < 4; ++j) {
      int n = l0 + wn + j * 16 + lo;
#pragma unroll
      for (int r = 0; r < 8; ++r) {
        int m = m0 + wm + i * 16 + hi * 8 + r;
        float v = acc[i][j][r] + b1[m];
        h[((long)b * 1024 + m) * 4096 + n] = f2bf(gelu_fast(v));
      }
    }
  }
}

// ---------------------------------------------------------------------------
// Kernel 4: GEMM2  out = x + (w2 @ h + b2)
//   A = w2 bf16 [512 x 1024], B = h bf16 [1024 x 4096] per batch
// grid = (16, 4, 16), block = 256
// ---------------------------------------------------------------------------
__global__ __launch_bounds__(256) void gemm2_res_kernel(
    const unsigned short* __restrict__ h,
    const unsigned short* __restrict__ w2b,
    const float* __restrict__ b2,
    const float* __restrict__ x,
    float* __restrict__ out) {
  __shared__ unsigned short ldsA[128 * BS];
  __shared__ unsigned short ldsB[256 * BS];
  const int tid  = threadIdx.x;
  const int lane = tid & 31, wave = tid >> 5;
  const int lo = lane & 15, hi = lane >> 4;
  const int wm = (wave & 1) * 64;
  const int wn = (wave >> 1) * 64;
  const int l0 = blockIdx.x * 256;
  const int m0 = blockIdx.y * 128;
  const int b  = blockIdx.z;
  const long hbase = (long)b * 1024 * 4096;

  v8f acc[4][4] = {};

  for (int k0 = 0; k0 < 1024; k0 += 32) {
    __syncthreads();
#pragma unroll
    for (int t = 0; t < 2; ++t) {
      int ch  = tid + t * 256;
      int row = ch >> 2;
      int kc  = (ch & 3) * 8;
      u32x4 v = *(const u32x4*)(w2b + (long)(m0 + row) * 1024 + k0 + kc);
      *(u32x4*)(&ldsA[row * BS + kc]) = v;
    }
    // --- B tile from bf16 h: channel-pair packed b32 stores ---
#pragma unroll
    for (int t = 0; t < 4; ++t) {
      int u  = tid + t * 256;
      int cp = u >> 6;
      int l  = (u & 63) * 4;
      const unsigned short* hp0 = h + hbase + (long)(k0 + 2 * cp) * 4096 + l0 + l;
      u32x2 r0 = *(const u32x2*)hp0;            // 4 bf16 of channel 2cp
      u32x2 r1 = *(const u32x2*)(hp0 + 4096);   // 4 bf16 of channel 2cp+1
      unsigned int* pb = (unsigned int*)&ldsB[l * BS + 2 * cp];
      pb[(0 * BS) / 2] = (r0.x & 0xFFFFu) | (r1.x << 16);
      pb[(1 * BS) / 2] = (r0.x >> 16)     | (r1.x & 0xFFFF0000u);
      pb[(2 * BS) / 2] = (r0.y & 0xFFFFu) | (r1.y << 16);
      pb[(3 * BS) / 2] = (r0.y >> 16)     | (r1.y & 0xFFFF0000u);
    }
    __syncthreads();
    Frag fb[4];
#pragma unroll
    for (int j = 0; j < 4; ++j) {
      const unsigned short* pb = &ldsB[(wn + j * 16 + lo) * BS + hi * 8];
      fb[j].q[0] = *(const u32x4*)(pb);
      fb[j].q[1] = *(const u32x4*)(pb + 16);
    }
#pragma unroll
    for (int i = 0; i < 4; ++i) {
      Frag fa;
      const unsigned short* pa = &ldsA[(wm + i * 16 + lo) * BS + hi * 8];
      fa.q[0] = *(const u32x4*)(pa);
      fa.q[1] = *(const u32x4*)(pa + 16);
#pragma unroll
      for (int j = 0; j < 4; ++j) {
        acc[i][j] = __builtin_amdgcn_wmma_f32_16x16x32_bf16(
            false, fa.v, false, fb[j].v, (short)0, acc[i][j], false, false);
      }
    }
  }
  // --- epilogue: + b2 + residual ---
#pragma unroll
  for (int i = 0; i < 4; ++i) {
#pragma unroll
    for (int j = 0; j < 4; ++j) {
      int n = l0 + wn + j * 16 + lo;
#pragma unroll
      for (int r = 0; r < 8; ++r) {
        int m = m0 + wm + i * 16 + hi * 8 + r;
        long idx = ((long)b * 512 + m) * 4096 + n;
        out[idx] = x[idx] + acc[i][j][r] + b2[m];
      }
    }
  }
}

// ---------------------------------------------------------------------------
// Launch: stats -> weight convert -> GEMM1(+GN+GELU) -> GEMM2(+residual)
// ws layout: [0,512) mean   [512,1024) rstd
//            [4096, +1MB) w1 bf16   [+1MB, +2MB) w2 bf16
//            [4096+2MB, +128MB) h bf16 [16,1024,4096]
// ---------------------------------------------------------------------------
extern "C" void kernel_launch(void* const* d_in, const int* in_sizes, int n_in,
                              void* d_out, int out_size, void* d_ws, size_t ws_size,
                              hipStream_t stream) {
  const float* x     = (const float*)d_in[0];
  const float* gamma = (const float*)d_in[1];
  const float* beta  = (const float*)d_in[2];
  const float* w1    = (const float*)d_in[3];
  const float* b1    = (const float*)d_in[4];
  const float* w2    = (const float*)d_in[5];
  const float* b2    = (const float*)d_in[6];
  float* out = (float*)d_out;

  char* ws = (char*)d_ws;
  float* mean          = (float*)(ws);
  float* rstd          = (float*)(ws + 512);
  unsigned short* w1b  = (unsigned short*)(ws + 4096);
  unsigned short* w2b  = (unsigned short*)(ws + 4096 + (1u << 20));
  unsigned short* h    = (unsigned short*)(ws + 4096 + (2u << 20));

  gn_stats_kernel<<<128, 256, 0, stream>>>(x, mean, rstd);
  cvt_w_kernel<<<2048, 256, 0, stream>>>(w1, w2, w1b, w2b);
  gemm1_gn_gelu_kernel<<<dim3(16, 8, 16), 256, 0, stream>>>(
      x, w1b, b1, gamma, beta, mean, rstd, h);
  gemm2_res_kernel<<<dim3(16, 4, 16), 256, 0, stream>>>(h, w2b, b2, x, out);
}
